// AttnDecoderRNN_19464791786192
// MI455X (gfx1250) — matrix operand order repaired and checked by reading, not compiled, not run on hardware
//
#include <hip/hip_runtime.h>
#include <hip/hip_bf16.h>

// ---------------------------------------------------------------------------
// AttnDecoderRNN single step for MI455X (gfx1250, wave32, WMMA).
// GEMMs in f16 with f32 accumulation via v_wmma_f32_16x16x32_f16.
// LDS staging via GLOBAL_LOAD_ASYNC_TO_LDS_B128 (ASYNCcnt-tracked DMA).
// ---------------------------------------------------------------------------

typedef __attribute__((ext_vector_type(16))) _Float16   v16h;
typedef __attribute__((ext_vector_type(8)))  float      v8f;
typedef __attribute__((ext_vector_type(4)))  unsigned   u32x4; // trivially-copyable 16B chunk

#if defined(__HIP_DEVICE_COMPILE__) && __has_builtin(__builtin_amdgcn_global_load_async_to_lds_b128)
#define HAVE_ASYNC_LDS 1
#define AS1 __attribute__((address_space(1)))
#define AS3 __attribute__((address_space(3)))
typedef int v4i __attribute__((vector_size(4 * sizeof(int)))); // matches builtin param type
#else
#define HAVE_ASYNC_LDS 0
#endif

#define Bsz 512
#define Hsz 512
#define Esz 300
#define LCsz 100
#define Osz 2004

// padded GEMM shapes
#define KA   52544   // attn_in 52524 -> pad to %32
#define NA   128     // LC 100 -> pad to %64
#define KX   1344    // gru_in 1324
#define NG   1536    // 3H
#define KM   1856    // mlp_in 1836
#define NO   2048    // O 2004
#define NZ1  1024    // 2H

#define BM 128
#define BN 64
#define BK 32
#define LDAH 40      // BK + 8 halves pad -> 80B row stride (16B multiple, bank-spread)

__device__ __forceinline__ float sigmf(float x) { return 1.0f / (1.0f + expf(-x)); }

union AFrag { v16h v; u32x4 q[2]; };

// ---------------------------------------------------------------------------
// Generic WMMA GEMM: C[M,N] = act(A[M,K] * W[N,K]^T + bias)
//   A,W f16 row-major (zero-padded so no bounds checks); C f32, optional f16
//   mirror.  gridDim = (N/64, M/128, Z). Z>1 => split-K partials via atomicAdd
//   (bias/act skipped; C must be pre-zeroed).
// act: 0=none, 1=tanh, 2=sigmoid
// ---------------------------------------------------------------------------
__global__ __launch_bounds__(256)
void wmma_gemm_f16(const _Float16* __restrict__ A, int lda,
                   const _Float16* __restrict__ W, int ldw,
                   float* __restrict__ C, int ldc,
                   _Float16* __restrict__ C16,
                   const float* __restrict__ bias, int nbias,
                   int K, int act)
{
    __shared__ __align__(16) _Float16 As[BM * LDAH];
    __shared__ __align__(16) _Float16 Ws[BN * LDAH];

    const int tid  = threadIdx.x;
    const int wave = tid >> 5;
    const int lane = tid & 31;
    const int wm   = wave >> 1;          // 0..3 (M direction)
    const int wn   = wave & 1;           // 0..1 (N direction)
    const int m_blk = blockIdx.y * BM;
    const int n_blk = blockIdx.x * BN;
    const int r16 = lane & 15;
    const int hf  = lane >> 4;

    // per-thread staging assignment (A: 2 chunks, W: 1 chunk of 16B)
    const int a_row0 = tid >> 2;               // 0..63
    const int a_row1 = (tid + 256) >> 2;       // 64..127
    const int a_col  = (tid & 3) << 3;         // halves: 0,8,16,24
    const int w_row  = tid >> 2;
    const int w_col  = (tid & 3) << 3;

    // split-K range
    const int nIters = (K + BK - 1) / BK;
    const int perZ   = (nIters + gridDim.z - 1) / gridDim.z;
    const int kbeg   = blockIdx.z * perZ * BK;
    int kend         = kbeg + perZ * BK;
    if (kend > K) kend = K;

    v8f acc[2][2] = {};

    for (int k0 = kbeg; k0 < kend; k0 += BK) {
#if HAVE_ASYNC_LDS
        // ---- CDNA5 async DMA: global -> LDS directly, tracked by ASYNCcnt
        __builtin_amdgcn_global_load_async_to_lds_b128(
            (AS1 v4i*)&A[(size_t)(m_blk + a_row0) * lda + k0 + a_col],
            (AS3 v4i*)&As[a_row0 * LDAH + a_col], 0, 0);
        __builtin_amdgcn_global_load_async_to_lds_b128(
            (AS1 v4i*)&A[(size_t)(m_blk + a_row1) * lda + k0 + a_col],
            (AS3 v4i*)&As[a_row1 * LDAH + a_col], 0, 0);
        __builtin_amdgcn_global_load_async_to_lds_b128(
            (AS1 v4i*)&W[(size_t)(n_blk + w_row) * ldw + k0 + w_col],
            (AS3 v4i*)&Ws[w_row * LDAH + w_col], 0, 0);
#if __has_builtin(__builtin_amdgcn_s_wait_asynccnt)
        __builtin_amdgcn_s_wait_asynccnt(0);
#else
        asm volatile("s_wait_asynccnt 0x0" ::: "memory");
#endif
#else
        // ---- fallback: global -> VGPR -> LDS
        *(u32x4*)&As[a_row0 * LDAH + a_col] =
            *(const u32x4*)&A[(size_t)(m_blk + a_row0) * lda + k0 + a_col];
        *(u32x4*)&As[a_row1 * LDAH + a_col] =
            *(const u32x4*)&A[(size_t)(m_blk + a_row1) * lda + k0 + a_col];
        *(u32x4*)&Ws[w_row * LDAH + w_col] =
            *(const u32x4*)&W[(size_t)(n_blk + w_row) * ldw + k0 + w_col];
        if (k0 + BK < kend) {  // prefetch next K tile (global_prefetch_b8)
            __builtin_prefetch(&A[(size_t)(m_blk + a_row0) * lda + k0 + BK + a_col], 0, 1);
            __builtin_prefetch(&W[(size_t)(n_blk + w_row) * ldw + k0 + BK + w_col], 0, 1);
        }
#endif
        __syncthreads();

        // ---- fragments per CDNA5 ISA layouts (CDNA5 ISA)
        AFrag a[2], b[2];
        #pragma unroll
        for (int i = 0; i < 2; ++i) {
            const int row = wm * 32 + i * 16 + r16;
            // A 16x32 f16: lanes<16 K=0..7 then 16..23 ; lanes>=16 K=8..15 then 24..31
            a[i].q[0] = *(const u32x4*)&As[row * LDAH + hf * 8];
            a[i].q[1] = *(const u32x4*)&As[row * LDAH + 16 + hf * 8];
        }
        #pragma unroll
        for (int j = 0; j < 2; ++j) {
            const int nrow = wn * 32 + j * 16 + r16;
            // B 32x16 f16: lanes<16 K=0..7 then 16..23 ; lanes>=16 K=8..15 then 24..31 (N = lane&15)
            b[j].q[0] = *(const u32x4*)&Ws[nrow * LDAH + hf * 8];
            b[j].q[1] = *(const u32x4*)&Ws[nrow * LDAH + 16 + hf * 8];
        }
        #pragma unroll
        for (int i = 0; i < 2; ++i)
            #pragma unroll
            for (int j = 0; j < 2; ++j)
                acc[i][j] = __builtin_amdgcn_wmma_f32_16x16x32_f16(
                    false, a[i].v, false, b[j].v, (short)0, acc[i][j], false, false);
        __syncthreads();
    }

    // ---- epilogue: C/D 16x16 f32: VGPR v -> row v + hf*8, col = lane&15
    const bool splitk = (gridDim.z > 1);
    #pragma unroll
    for (int i = 0; i < 2; ++i) {
        #pragma unroll
        for (int j = 0; j < 2; ++j) {
            const int n  = n_blk + wn * 32 + j * 16 + r16;
            const float bv = (!splitk && bias && n < nbias) ? bias[n] : 0.0f;
            #pragma unroll
            for (int v = 0; v < 8; ++v) {
                const int m = m_blk + wm * 32 + i * 16 + hf * 8 + v;
                float x = acc[i][j][v];
                if (splitk) {
                    atomicAdd(&C[(size_t)m * ldc + n], x);
                } else {
                    x += bv;
                    if (act == 1)      x = tanhf(x);
                    else if (act == 2) x = sigmf(x);
                    C[(size_t)m * ldc + n] = x;
                    if (C16) C16[(size_t)m * ldc + n] = (_Float16)x;
                }
            }
        }
    }
}

// ---------------------------------------------------------------------------
// helpers
// ---------------------------------------------------------------------------
__global__ void zero_f32(float* p, int n) {
    int i = blockIdx.x * blockDim.x + threadIdx.x;
    if (i < n) p[i] = 0.0f;
}

// f32 [N,K] -> f16 [Np,Kp] zero-padded
__global__ void convert_w_f16(const float* __restrict__ W, int N, int K,
                              _Float16* __restrict__ out, int Np, int Kp) {
    size_t idx = (size_t)blockIdx.x * blockDim.x + threadIdx.x;
    if (idx >= (size_t)Np * Kp) return;
    int n = (int)(idx / Kp), k = (int)(idx % Kp);
    out[idx] = (n < N && k < K) ? (_Float16)W[(size_t)n * K + k] : (_Float16)0.0f;
}

__global__ void embed_kernel(const int* __restrict__ input, const int* __restrict__ input_d,
                             const float* __restrict__ emb_w, const float* __restrict__ emb_d_w,
                             float* __restrict__ emb) {
    int idx = blockIdx.x * blockDim.x + threadIdx.x;
    if (idx >= Bsz * Esz) return;
    int b = idx / Esz, e = idx % Esz;
    int tok = input[b];
    float v;
    if (tok > Osz - 1) {                       // doc pointer
        int di = tok - Osz;
        di = di < 0 ? 0 : (di > LCsz - 1 ? LCsz - 1 : di);
        int dt = input_d[b * LCsz + di];
        v = emb_d_w[(size_t)dt * Esz + e];
    } else {
        int ct = tok < 0 ? 0 : (tok > 2103 ? 2103 : tok);
        v = emb_w[(size_t)ct * Esz + e];
    }
    emb[idx] = v;
}

// in_attn = [h0 | emb | ans | annot.flat] -> f16, padded to KA
__global__ void pack_attn(const float* __restrict__ h0, const float* __restrict__ emb,
                          const float* __restrict__ ans, const float* __restrict__ annot,
                          _Float16* __restrict__ A) {
    size_t idx = (size_t)blockIdx.x * blockDim.x + threadIdx.x;
    if (idx >= (size_t)Bsz * KA) return;
    int b = (int)(idx / KA), k = (int)(idx % KA);
    float v = 0.0f;
    if (k < Hsz)              v = h0[b * Hsz + k];
    else if (k < Hsz + Esz)   v = emb[b * Esz + (k - Hsz)];
    else if (k < 2*Hsz + Esz) v = ans[b * Hsz + (k - Hsz - Esz)];
    else if (k < 2*Hsz + Esz + LCsz * Hsz)
        v = annot[(size_t)b * (LCsz * Hsz) + (k - 2*Hsz - Esz)];
    A[idx] = (_Float16)v;
}

// tanh + softmax over LC (tanh is bounded -> direct exp is safe)
__global__ void attn_softmax(const float* __restrict__ logits, // [B, NA]
                             const float* __restrict__ attn_b,
                             float* __restrict__ aw) {          // [B, LC]
    __shared__ float red[128];
    int b = blockIdx.x, t = threadIdx.x;
    float e = 0.0f;
    if (t < LCsz) e = expf(tanhf(logits[(size_t)b * NA + t] + attn_b[t]));
    red[t] = e; __syncthreads();
    for (int s = 64; s > 0; s >>= 1) { if (t < s) red[t] += red[t + s]; __syncthreads(); }
    if (t < LCsz) aw[b * LCsz + t] = e / red[0];
}

// c_t[b,h] = sum_l aw[b,l] * annot[b,l,h]
__global__ void ctx_kernel(const float* __restrict__ aw, const float* __restrict__ annot,
                           float* __restrict__ ct) {
    int idx = blockIdx.x * blockDim.x + threadIdx.x;
    if (idx >= Bsz * Hsz) return;
    int b = idx >> 9, h = idx & (Hsz - 1);
    const float* an = annot + (size_t)b * (LCsz * Hsz) + h;
    const float* w  = aw + b * LCsz;
    float s = 0.0f;
    #pragma unroll 4
    for (int l = 0; l < LCsz; ++l) s += w[l] * an[(size_t)l * Hsz];
    ct[idx] = s;
}

// x = [c_t | ans | emb] f16 padded to KX
__global__ void pack_x(const float* __restrict__ ct, const float* __restrict__ ans,
                       const float* __restrict__ emb, _Float16* __restrict__ X) {
    size_t idx = (size_t)blockIdx.x * blockDim.x + threadIdx.x;
    if (idx >= (size_t)Bsz * KX) return;
    int b = (int)(idx / KX), k = (int)(idx % KX);
    float v = 0.0f;
    if (k < Hsz)                v = ct[b * Hsz + k];
    else if (k < 2*Hsz)         v = ans[b * Hsz + (k - Hsz)];
    else if (k < 2*Hsz + Esz)   v = emb[b * Esz + (k - 2*Hsz)];
    X[idx] = (_Float16)v;
}

// GRU gate fusion; writes h_new (ws) and both h_new output copies.
__global__ void gru_kernel(const float* __restrict__ gi, const float* __restrict__ gh,
                           const float* __restrict__ h0, float* __restrict__ hnew,
                           float* __restrict__ out) {
    int idx = blockIdx.x * blockDim.x + threadIdx.x;
    if (idx >= Bsz * Hsz) return;
    int b = idx >> 9, h = idx & (Hsz - 1);
    const float* gib = gi + (size_t)b * NG;
    const float* ghb = gh + (size_t)b * NG;
    float r = sigmf(gib[h]           + ghb[h]);
    float z = sigmf(gib[Hsz + h]     + ghb[Hsz + h]);
    float n = tanhf(gib[2*Hsz + h] + r * ghb[2*Hsz + h]);
    float hv = (1.0f - z) * n + z * h0[idx];
    hnew[idx] = hv;
    out[idx] = hv;                 // h_new[:,None,:]  (B,1,H) flat
    out[Bsz * Hsz + idx] = hv;     // h_new[None,:,:]  (1,B,H) flat (same layout)
}

// input_mlp = [h_new | c_t | ans | emb] f16 padded to KM
__global__ void pack_mlp(const float* __restrict__ hn, const float* __restrict__ ct,
                         const float* __restrict__ ans, const float* __restrict__ emb,
                         _Float16* __restrict__ X) {
    size_t idx = (size_t)blockIdx.x * blockDim.x + threadIdx.x;
    if (idx >= (size_t)Bsz * KM) return;
    int b = (int)(idx / KM), k = (int)(idx % KM);
    float v = 0.0f;
    if (k < Hsz)                v = hn[b * Hsz + k];
    else if (k < 2*Hsz)         v = ct[b * Hsz + (k - Hsz)];
    else if (k < 3*Hsz)         v = ans[b * Hsz + (k - 2*Hsz)];
    else if (k < 3*Hsz + Esz)   v = emb[b * Esz + (k - 3*Hsz)];
    X[idx] = (_Float16)v;
}

// z_t[b] = sigmoid(z2[b,:] . zt_W3 + zt_b3)
__global__ void zt_kernel(const float* __restrict__ z2, const float* __restrict__ w3,
                          const float* __restrict__ b3, float* __restrict__ zt) {
    int b = blockIdx.x * blockDim.x + threadIdx.x;
    if (b >= Bsz) return;
    const float* r = z2 + (size_t)b * Hsz;
    float s = b3[0];
    #pragma unroll 4
    for (int k = 0; k < Hsz; ++k) s += r[k] * w3[k];
    zt[b] = sigmf(s);
}

// softmax over O logits (bias already fused) + p_t = log-concat, one block/row
__global__ void out_softmax_pt(const float* __restrict__ ologits, // [B, NO]
                               const float* __restrict__ aw,      // [B, LC]
                               const float* __restrict__ zt,      // [B]
                               float* __restrict__ pt) {          // [B, O+LC]
    __shared__ float red[256];
    int b = blockIdx.x, t = threadIdx.x;
    const float* row = ologits + (size_t)b * NO;
    float m = -1e30f;
    for (int n = t; n < Osz; n += 256) m = fmaxf(m, row[n]);
    red[t] = m; __syncthreads();
    for (int s = 128; s > 0; s >>= 1) { if (t < s) red[t] = fmaxf(red[t], red[t + s]); __syncthreads(); }
    m = red[0]; __syncthreads();
    float ssum = 0.0f;
    for (int n = t; n < Osz; n += 256) ssum += expf(row[n] - m);
    red[t] = ssum; __syncthreads();
    for (int s = 128; s > 0; s >>= 1) { if (t < s) red[t] += red[t + s]; __syncthreads(); }
    float inv = 1.0f / red[0];
    float z = zt[b];
    float* out = pt + (size_t)b * (Osz + LCsz);
    for (int n = t; n < Osz; n += 256)
        out[n] = logf(expf(row[n] - m) * inv * z + 1e-20f);
    for (int l = t; l < LCsz; l += 256)
        out[Osz + l] = logf(aw[b * LCsz + l] * (1.0f - z) + 1e-20f);
}

// ---------------------------------------------------------------------------
extern "C" void kernel_launch(void* const* d_in, const int* in_sizes, int n_in,
                              void* d_out, int out_size, void* d_ws, size_t ws_size,
                              hipStream_t stream) {
    (void)in_sizes; (void)n_in; (void)out_size; (void)ws_size;

    const int*   input   = (const int*)  d_in[0];
    const int*   input_d = (const int*)  d_in[1];
    const float* hidden  = (const float*)d_in[2];   // (1,B,H)
    const float* annot   = (const float*)d_in[3];   // (B,LC,H)
    const float* ans     = (const float*)d_in[4];   // (B,1,H)
    const float* emb_w   = (const float*)d_in[5];
    const float* emb_d_w = (const float*)d_in[6];
    const float* attn_W  = (const float*)d_in[7];
    const float* attn_b  = (const float*)d_in[8];
    const float* W_ih    = (const float*)d_in[9];
    const float* W_hh    = (const float*)d_in[10];
    const float* b_ih    = (const float*)d_in[11];
    const float* b_hh    = (const float*)d_in[12];
    const float* mlp_W   = (const float*)d_in[13];
    const float* mlp_b   = (const float*)d_in[14];
    const float* out_W   = (const float*)d_in[15];
    const float* out_b   = (const float*)d_in[16];
    const float* zt_W1   = (const float*)d_in[17];
    const float* zt_b1   = (const float*)d_in[18];
    const float* zt_W2   = (const float*)d_in[19];
    const float* zt_b2   = (const float*)d_in[20];
    const float* zt_W3   = (const float*)d_in[21];
    const float* zt_b3   = (const float*)d_in[22];
    float* out = (float*)d_out;

    // ---- workspace bump allocator (256B aligned)
    char* wsb = (char*)d_ws;
    size_t off = 0;
    auto alloc = [&](size_t bytes) -> void* {
        void* p = wsb + off;
        off = (off + bytes + 255) & ~(size_t)255;
        return p;
    };
    float*     emb     = (float*)    alloc((size_t)Bsz * Esz * 4);
    _Float16*  Aattn   = (_Float16*) alloc((size_t)Bsz * KA * 2);
    _Float16*  Wattn   = (_Float16*) alloc((size_t)NA * KA * 2);
    float*     alog    = (float*)    alloc((size_t)Bsz * NA * 4);
    float*     aw      = (float*)    alloc((size_t)Bsz * LCsz * 4);
    float*     ct      = (float*)    alloc((size_t)Bsz * Hsz * 4);
    _Float16*  Xgru    = (_Float16*) alloc((size_t)Bsz * KX * 2);
    _Float16*  Wih     = (_Float16*) alloc((size_t)NG * KX * 2);
    _Float16*  h0f16   = (_Float16*) alloc((size_t)Bsz * Hsz * 2);
    _Float16*  Whh     = (_Float16*) alloc((size_t)NG * Hsz * 2);
    float*     gi      = (float*)    alloc((size_t)Bsz * NG * 4);
    float*     gh      = (float*)    alloc((size_t)Bsz * NG * 4);
    float*     hnew    = (float*)    alloc((size_t)Bsz * Hsz * 4);
    _Float16*  Xmlp    = (_Float16*) alloc((size_t)Bsz * KM * 2);
    _Float16*  Wmlp    = (_Float16*) alloc((size_t)Hsz * KM * 2);
    float*     et      = (float*)    alloc((size_t)Bsz * Hsz * 4);
    _Float16*  et16    = (_Float16*) alloc((size_t)Bsz * Hsz * 2);
    _Float16*  Wout    = (_Float16*) alloc((size_t)NO * Hsz * 2);
    float*     olog    = (float*)    alloc((size_t)Bsz * NO * 4);
    _Float16*  Wz1     = (_Float16*) alloc((size_t)NZ1 * KM * 2);
    float*     z1      = (float*)    alloc((size_t)Bsz * NZ1 * 4);
    _Float16*  z1f16   = (_Float16*) alloc((size_t)Bsz * NZ1 * 2);
    _Float16*  Wz2     = (_Float16*) alloc((size_t)Hsz * NZ1 * 2);
    float*     z2      = (float*)    alloc((size_t)Bsz * Hsz * 4);
    float*     ztv     = (float*)    alloc((size_t)Bsz * 4);

    const int TB = 256;
    auto blk1 = [](size_t n, int tb) { return (unsigned)((n + tb - 1) / tb); };

    // ---- 1. weight conversions (f32 -> zero-padded f16)
    convert_w_f16<<<blk1((size_t)NA * KA, TB), TB, 0, stream>>>(attn_W, LCsz, 52524, Wattn, NA, KA);
    convert_w_f16<<<blk1((size_t)NG * KX, TB), TB, 0, stream>>>(W_ih, NG, 1324, Wih, NG, KX);
    convert_w_f16<<<blk1((size_t)NG * Hsz, TB), TB, 0, stream>>>(W_hh, NG, Hsz, Whh, NG, Hsz);
    convert_w_f16<<<blk1((size_t)Hsz * KM, TB), TB, 0, stream>>>(mlp_W, Hsz, 1836, Wmlp, Hsz, KM);
    convert_w_f16<<<blk1((size_t)NO * Hsz, TB), TB, 0, stream>>>(out_W, Osz, Hsz, Wout, NO, Hsz);
    convert_w_f16<<<blk1((size_t)NZ1 * KM, TB), TB, 0, stream>>>(zt_W1, NZ1, 1836, Wz1, NZ1, KM);
    convert_w_f16<<<blk1((size_t)Hsz * NZ1, TB), TB, 0, stream>>>(zt_W2, Hsz, NZ1, Wz2, Hsz, NZ1);
    convert_w_f16<<<blk1((size_t)Bsz * Hsz, TB), TB, 0, stream>>>(hidden, Bsz, Hsz, h0f16, Bsz, Hsz);

    // ---- 2. embedding + attention input pack
    embed_kernel<<<blk1(Bsz * Esz, TB), TB, 0, stream>>>(input, input_d, emb_w, emb_d_w, emb);
    pack_attn<<<blk1((size_t)Bsz * KA, TB), TB, 0, stream>>>(hidden, emb, ans, annot, Aattn);

    // ---- 3. attention GEMM (split-K=8 for occupancy) + softmax
    zero_f32<<<blk1(Bsz * NA, TB), TB, 0, stream>>>(alog, Bsz * NA);
    wmma_gemm_f16<<<dim3(NA / BN, Bsz / BM, 8), TB, 0, stream>>>(
        Aattn, KA, Wattn, KA, alog, NA, nullptr, nullptr, 0, KA, 0);
    attn_softmax<<<Bsz, 128, 0, stream>>>(alog, attn_b, aw);

    // ---- 4. context + GRU
    ctx_kernel<<<blk1(Bsz * Hsz, TB), TB, 0, stream>>>(aw, annot, ct);
    pack_x<<<blk1((size_t)Bsz * KX, TB), TB, 0, stream>>>(ct, ans, emb, Xgru);
    wmma_gemm_f16<<<dim3(NG / BN, Bsz / BM, 1), TB, 0, stream>>>(
        Xgru, KX, Wih, KX, gi, NG, nullptr, b_ih, NG, KX, 0);
    wmma_gemm_f16<<<dim3(NG / BN, Bsz / BM, 1), TB, 0, stream>>>(
        h0f16, Hsz, Whh, Hsz, gh, NG, nullptr, b_hh, NG, Hsz, 0);
    gru_kernel<<<blk1(Bsz * Hsz, TB), TB, 0, stream>>>(gi, gh, hidden, hnew, out);

    // ---- 5. mlp -> output logits ; z-gate chain
    pack_mlp<<<blk1((size_t)Bsz * KM, TB), TB, 0, stream>>>(hnew, ct, ans, emb, Xmlp);
    wmma_gemm_f16<<<dim3(Hsz / BN, Bsz / BM, 1), TB, 0, stream>>>(
        Xmlp, KM, Wmlp, KM, et, Hsz, et16, mlp_b, Hsz, KM, 0);
    wmma_gemm_f16<<<dim3(NO / BN, Bsz / BM, 1), TB, 0, stream>>>(
        et16, Hsz, Wout, Hsz, olog, NO, nullptr, out_b, Osz, Hsz, 0);
    wmma_gemm_f16<<<dim3(NZ1 / BN, Bsz / BM, 1), TB, 0, stream>>>(
        Xmlp, KM, Wz1, KM, z1, NZ1, z1f16, zt_b1, NZ1, KM, 1 /*tanh*/);
    wmma_gemm_f16<<<dim3(Hsz / BN, Bsz / BM, 1), TB, 0, stream>>>(
        z1f16, NZ1, Wz2, NZ1, z2, Hsz, nullptr, zt_b2, Hsz, NZ1, 1 /*tanh*/);
    zt_kernel<<<blk1(Bsz, TB), TB, 0, stream>>>(z2, zt_W3, zt_b3, ztv);

    // ---- 6. fused output softmax + log-concat p_t
    out_softmax_pt<<<Bsz, 256, 0, stream>>>(olog, aw, ztv, out + (size_t)2 * Bsz * Hsz);
}
